// ActorCriticRNN_65438121722400
// MI455X (gfx1250) — matrix operand order repaired and checked by reading, not compiled, not run on hardware
//
#include <hip/hip_runtime.h>
#include <hip/hip_bf16.h>
#include <stdint.h>

#define TT 512
#define BB 256
#define OBS_D 128
#define HH 256
#define AA 32
#define TB (TT * BB)

typedef __attribute__((ext_vector_type(8))) float v8f;
typedef __attribute__((ext_vector_type(16))) __bf16 v16bf;

// ---------------- fragment helpers (CDNA5 WMMA 16x16x32 bf16) ----------------
// A (16x32, MxK): lane L holds row M = L&15.
//   lanes 0-15 : vec[0..7] = K kb+0..7, vec[8..15] = K kb+16..23, kb=0
//   lanes 16-31: same with kb=8
__device__ __forceinline__ v16bf a_frag_f32(const float* __restrict__ p) {
  v16bf f;
#pragma unroll
  for (int i = 0; i < 8; ++i) f[i] = (__bf16)p[i];
#pragma unroll
  for (int i = 0; i < 8; ++i) f[8 + i] = (__bf16)p[16 + i];
  return f;
}

__device__ __forceinline__ v16bf a_frag_bf16(const __bf16* p) {
  union { uint4 u[2]; v16bf v; } t;
  const uint4* q = (const uint4*)p;
  t.u[0] = q[0];   // K kb+0..7
  t.u[1] = q[2];   // K kb+16..23
  return t.v;
}

// B (32x16, KxN): lane L holds col N = L&15; same K order as A (kb=0 / kb=8)
__device__ __forceinline__ v16bf b_frag_bf16(const __bf16* p) {
  union { uint4 u[2]; v16bf v; } t;
  const uint4* q = (const uint4*)p;
  t.u[0] = q[0];
  t.u[1] = q[2];
  return t.v;
}

__device__ __forceinline__ v8f wmma_bf16(v16bf a, v16bf b, v8f c) {
  return __builtin_amdgcn_wmma_f32_16x16x32_bf16(false, a, false, b, (short)0, c,
                                                 false, false);
}

__device__ __forceinline__ float sigmoidf_(float x) { return 1.f / (1.f + __expf(-x)); }
__device__ __forceinline__ float tanhf_(float x) {
  float e = __expf(-2.f * fabsf(x));
  float r = (1.f - e) / (1.f + e);
  return x < 0.f ? -r : r;
}

// ---------------- weight prep: f32 [K][N] -> bf16 [Npad][K] (zero pad) -------
__global__ void k_transpose_bf16(const float* __restrict__ src, __bf16* __restrict__ dst,
                                 int K, int N, int Npad) {
  int idx = blockIdx.x * blockDim.x + threadIdx.x;
  if (idx >= Npad * K) return;
  int n = idx / K, k = idx % K;
  float v = (n < N) ? src[(size_t)k * N + n] : 0.f;
  dst[(size_t)n * K + k] = (__bf16)v;
}

// ---------------- init: zero grid barrier, seed h buffers --------------------
__global__ void k_init(unsigned* __restrict__ bar, float* __restrict__ hbuf0,
                       __bf16* __restrict__ hbf0, const float* __restrict__ hidden) {
  int idx = blockIdx.x * blockDim.x + threadIdx.x;
  if (idx == 0) *bar = 0u;
  if (idx < BB * HH) {
    float v = hidden[idx];
    hbuf0[idx] = v;
    hbf0[idx] = (__bf16)v;
  }
}

// ---------------- emb = relu(obs @ W_emb + b_emb), bf16 out ------------------
// one wave per 16-row block; A streamed from HBM once, all 16 N tiles reuse it
__global__ void k_emb(const float* __restrict__ obs, const __bf16* __restrict__ Wembt,
                      const float* __restrict__ b_emb, __bf16* __restrict__ emb) {
  int rb = blockIdx.x * 8 + (threadIdx.x >> 5);
  int m0 = rb * 16;
  int lane = threadIdx.x & 31;
  int fr = lane & 15;
  int kbA = (lane & 16) ? 8 : 0;
  int kbB = (lane & 16) ? 8 : 0;
  int row = m0 + fr;
  int c = lane & 15, rhalf = (lane & 16) ? 8 : 0;

  v16bf af[4];
#pragma unroll
  for (int kk = 0; kk < 4; ++kk)
    af[kk] = a_frag_f32(obs + (size_t)row * OBS_D + kk * 32 + kbA);

  for (int tn = 0; tn < 16; ++tn) {
    v8f acc = {};
#pragma unroll
    for (int kk = 0; kk < 4; ++kk) {
      v16bf b = b_frag_bf16(Wembt + (size_t)(tn * 16 + fr) * OBS_D + kk * 32 + kbB);
      acc = wmma_bf16(af[kk], b, acc);
    }
    int n = tn * 16 + c;
    float bias = b_emb[n];
#pragma unroll
    for (int i = 0; i < 8; ++i) {
      float v = acc[i] + bias;
      emb[(size_t)(m0 + i + rhalf) * HH + n] = (__bf16)(v > 0.f ? v : 0.f);
    }
  }
}

// ---------------- gi = emb @ Wi + bi, f32 out [TB][768] ----------------------
// one wave per 16-row block; 48 N tiles, 384 WMMA per wave
__global__ void k_gi(const __bf16* __restrict__ emb, const __bf16* __restrict__ Wit,
                     const float* __restrict__ bi, float* __restrict__ gi) {
  int rb = blockIdx.x * 8 + (threadIdx.x >> 5);
  int m0 = rb * 16;
  int lane = threadIdx.x & 31;
  int fr = lane & 15;
  int kbA = (lane & 16) ? 8 : 0;
  int kbB = (lane & 16) ? 8 : 0;
  int row = m0 + fr;
  int c = lane & 15, rhalf = (lane & 16) ? 8 : 0;

  v16bf af[8];
#pragma unroll
  for (int kk = 0; kk < 8; ++kk)
    af[kk] = a_frag_bf16(emb + (size_t)row * HH + kk * 32 + kbA);

  for (int tn = 0; tn < 48; ++tn) {
    v8f acc = {};
#pragma unroll
    for (int kk = 0; kk < 8; ++kk) {
      v16bf b = b_frag_bf16(Wit + (size_t)(tn * 16 + fr) * HH + kk * 32 + kbB);
      acc = wmma_bf16(af[kk], b, acc);
    }
    int n = tn * 16 + c;
    float bias = bi[n];
#pragma unroll
    for (int i = 0; i < 8; ++i)
      gi[(size_t)(m0 + i + rhalf) * (3 * HH) + n] = acc[i] + bias;
  }
}

// ---------------- persistent GRU scan (32 blocks, grid barrier per step) -----
__device__ __forceinline__ void grid_barrier(unsigned* cnt, unsigned target) {
  __threadfence();
  __syncthreads();
  if (threadIdx.x == 0) {
    __hip_atomic_fetch_add(cnt, 1u, __ATOMIC_RELEASE, __HIP_MEMORY_SCOPE_AGENT);
    while (__hip_atomic_load(cnt, __ATOMIC_ACQUIRE, __HIP_MEMORY_SCOPE_AGENT) < target)
      __builtin_amdgcn_s_sleep(2);
  }
  __syncthreads();
  __threadfence();
}

__global__ void __launch_bounds__(256, 1)
k_gru(float* __restrict__ hbuf, __bf16* __restrict__ hbf,
      const float* __restrict__ gi, const unsigned char* __restrict__ dones,
      const __bf16* __restrict__ Wht, const float* __restrict__ bhn,
      __bf16* __restrict__ y, float* __restrict__ hidden_out,
      unsigned* __restrict__ bar) {
  // block pair (2k,2k+1) owns hidden-column slab j0 = k*16; weights cached in LDS
  __shared__ __align__(16) __bf16 wslab[3 * 16 * 256];  // 24KB: [gate][col][k]
  int j0 = (blockIdx.x >> 1) * 16;
  for (int idx = threadIdx.x; idx < 3 * 16 * 256; idx += 256) {
    int g = idx >> 12, rem = idx & 4095, cl = rem >> 8, k = rem & 255;
    wslab[idx] = Wht[(size_t)g * (HH * HH) + (size_t)(j0 + cl) * HH + k];
  }
  __syncthreads();

  int lw = threadIdx.x >> 5;
  int w = blockIdx.x * 8 + lw;
  int m0 = (w & 15) * 16;          // batch tile
  int lane = threadIdx.x & 31;
  int fr = lane & 15;
  int kbA = (lane & 16) ? 8 : 0;
  int kbB = (lane & 16) ? 8 : 0;
  int brow = m0 + fr;
  int c = lane & 15, rhalf = (lane & 16) ? 8 : 0;
  unsigned nblk = gridDim.x;

  // prefetch gi/dones for t=0 (streamed from HBM; hide the first-step latency)
  {
    const float* g0 = gi + (size_t)brow * (3 * HH) + j0;
    __builtin_prefetch(g0 + ((lane & 16) ? HH : 0), 0, 1);
    if (lane < 16) __builtin_prefetch(g0 + 2 * HH, 0, 1);
    if (lane == 0) __builtin_prefetch(dones, 0, 1);
  }

  for (int t = 0; t < TT; ++t) {
    const float* h_in = hbuf + (size_t)(t & 1) * (BB * HH);
    float* h_out = hbuf + (size_t)((t + 1) & 1) * (BB * HH);
    const __bf16* hb_in = hbf + (size_t)(t & 1) * (BB * HH);
    __bf16* hb_out = hbf + (size_t)((t + 1) & 1) * (BB * HH);

    // GEMM on UNMASKED h (bf16 shadow): done-mask applied exactly in epilogue,
    // since WMMA output row r depends only on A row r.
    v8f ar = {}, az = {}, an = {};
#pragma unroll
    for (int k0 = 0; k0 < HH; k0 += 32) {
      v16bf a = a_frag_bf16(hb_in + (size_t)brow * HH + k0 + kbA);
      v16bf br = b_frag_bf16(wslab + 0 * 4096 + fr * 256 + k0 + kbB);
      v16bf bz = b_frag_bf16(wslab + 1 * 4096 + fr * 256 + k0 + kbB);
      v16bf bn = b_frag_bf16(wslab + 2 * 4096 + fr * 256 + k0 + kbB);
      ar = wmma_bf16(a, br, ar);
      az = wmma_bf16(a, bz, az);
      an = wmma_bf16(a, bn, an);
    }

    int j = j0 + c;
    float bh = bhn[j];
#pragma unroll
    for (int i = 0; i < 8; ++i) {
      int r = m0 + i + rhalf;
      const float* gp = gi + ((size_t)t * BB + r) * (3 * HH);
      bool dn = dones[(size_t)t * BB + r] != 0;
      float hprev = dn ? 0.f : h_in[(size_t)r * HH + j];
      float arm = dn ? 0.f : ar[i];
      float azm = dn ? 0.f : az[i];
      float anm = dn ? 0.f : an[i];
      float rr = sigmoidf_(gp[j] + arm);
      float zz = sigmoidf_(gp[HH + j] + azm);
      float nn = tanhf_(gp[2 * HH + j] + rr * (anm + bh));
      float hnew = (1.f - zz) * nn + zz * hprev;
      h_out[(size_t)r * HH + j] = hnew;
      hb_out[(size_t)r * HH + j] = (__bf16)hnew;
      y[((size_t)t * BB + r) * HH + j] = (__bf16)hnew;
      if (t == TT - 1) hidden_out[(size_t)r * HH + j] = hnew;
    }

    // prefetch next step's gi rows + done flags; overlaps with the barrier wait
    if (t + 1 < TT) {
      const float* gn = gi + ((size_t)(t + 1) * BB + brow) * (3 * HH) + j0;
      __builtin_prefetch(gn + ((lane & 16) ? HH : 0), 0, 1);   // gate r / z rows
      if (lane < 16) __builtin_prefetch(gn + 2 * HH, 0, 1);    // gate n rows
      if (lane == 0) __builtin_prefetch(dones + (size_t)(t + 1) * BB, 0, 1);
    }
    grid_barrier(bar, nblk * (unsigned)(t + 1));
  }
}

// ---------------- actor/critic heads: one wave per 16-row block --------------
__global__ void k_heads(const __bf16* __restrict__ y,
                        const __bf16* __restrict__ Wa1t, const float* __restrict__ b_a1,
                        const __bf16* __restrict__ Wa2t, const float* __restrict__ b_a2,
                        const __bf16* __restrict__ Wc1t, const float* __restrict__ b_c1,
                        const __bf16* __restrict__ Wc2t, const float* __restrict__ b_c2,
                        const float* __restrict__ avail,
                        float* __restrict__ logits, float* __restrict__ value) {
  __shared__ __align__(16) __bf16 hl[8][16 * 256];  // 8KB per wave
  int lw = threadIdx.x >> 5;
  int rb = blockIdx.x * 8 + lw;
  int m0 = rb * 16;
  int lane = threadIdx.x & 31;
  int fr = lane & 15;
  int kbA = (lane & 16) ? 8 : 0;
  int kbB = (lane & 16) ? 8 : 0;
  int row = m0 + fr;
  int c = lane & 15, rhalf = (lane & 16) ? 8 : 0;
  __bf16* buf = hl[lw];

  v16bf af[8];
#pragma unroll
  for (int kk = 0; kk < 8; ++kk)
    af[kk] = a_frag_bf16(y + (size_t)row * HH + kk * 32 + kbA);

  // actor hidden -> LDS (bf16)
  for (int tn = 0; tn < 16; ++tn) {
    v8f acc = {};
#pragma unroll
    for (int kk = 0; kk < 8; ++kk) {
      v16bf b = b_frag_bf16(Wa1t + (size_t)(tn * 16 + fr) * HH + kk * 32 + kbB);
      acc = wmma_bf16(af[kk], b, acc);
    }
    int n = tn * 16 + c;
    float bias = b_a1[n];
#pragma unroll
    for (int i = 0; i < 8; ++i) {
      float v = acc[i] + bias;
      buf[(i + rhalf) * 256 + n] = (__bf16)(v > 0.f ? v : 0.f);
    }
  }
  // logits = actor_h @ W_a2 + b_a2 - (1-avail)*1e10
  for (int tn = 0; tn < 2; ++tn) {
    v8f acc = {};
#pragma unroll
    for (int kk = 0; kk < 8; ++kk) {
      v16bf a = a_frag_bf16(buf + fr * 256 + kk * 32 + kbA);
      v16bf b = b_frag_bf16(Wa2t + (size_t)(tn * 16 + fr) * HH + kk * 32 + kbB);
      acc = wmma_bf16(a, b, acc);
    }
    int n = tn * 16 + c;
    float bias = b_a2[n];
#pragma unroll
    for (int i = 0; i < 8; ++i) {
      size_t r = (size_t)(m0 + i + rhalf);
      float am = acc[i] + bias;
      float av = avail[r * AA + n];
      logits[r * AA + n] = am - (1.f - av) * 1e10f;
    }
  }
  // critic hidden -> LDS (overwrite; same-wave DS ops are in-order)
  for (int tn = 0; tn < 16; ++tn) {
    v8f acc = {};
#pragma unroll
    for (int kk = 0; kk < 8; ++kk) {
      v16bf b = b_frag_bf16(Wc1t + (size_t)(tn * 16 + fr) * HH + kk * 32 + kbB);
      acc = wmma_bf16(af[kk], b, acc);
    }
    int n = tn * 16 + c;
    float bias = b_c1[n];
#pragma unroll
    for (int i = 0; i < 8; ++i) {
      float v = acc[i] + bias;
      buf[(i + rhalf) * 256 + n] = (__bf16)(v > 0.f ? v : 0.f);
    }
  }
  // value = critic_h @ W_c2 (padded to 16 cols, only col 0 real) + b_c2
  {
    v8f acc = {};
#pragma unroll
    for (int kk = 0; kk < 8; ++kk) {
      v16bf a = a_frag_bf16(buf + fr * 256 + kk * 32 + kbA);
      v16bf b = b_frag_bf16(Wc2t + (size_t)fr * HH + kk * 32 + kbB);
      acc = wmma_bf16(a, b, acc);
    }
    if (c == 0) {
      float bias = b_c2[0];
#pragma unroll
      for (int i = 0; i < 8; ++i)
        value[m0 + i + rhalf] = acc[i] + bias;
    }
  }
}

// ---------------- host side --------------------------------------------------
extern "C" void kernel_launch(void* const* d_in, const int* in_sizes, int n_in,
                              void* d_out, int out_size, void* d_ws, size_t ws_size,
                              hipStream_t stream) {
  (void)in_sizes; (void)n_in; (void)out_size; (void)ws_size;
  const float* hidden = (const float*)d_in[0];
  const float* obs    = (const float*)d_in[1];
  const unsigned char* dones = (const unsigned char*)d_in[2];
  const float* avail  = (const float*)d_in[3];
  const float* W_emb = (const float*)d_in[4];
  const float* b_emb = (const float*)d_in[5];
  const float* Wi    = (const float*)d_in[6];
  const float* bi    = (const float*)d_in[7];
  const float* Whr   = (const float*)d_in[8];
  const float* Whz   = (const float*)d_in[9];
  const float* Whn   = (const float*)d_in[10];
  const float* bhn   = (const float*)d_in[11];
  const float* W_a1  = (const float*)d_in[12];
  const float* b_a1  = (const float*)d_in[13];
  const float* W_a2  = (const float*)d_in[14];
  const float* b_a2  = (const float*)d_in[15];
  const float* W_c1  = (const float*)d_in[16];
  const float* b_c1  = (const float*)d_in[17];
  const float* W_c2  = (const float*)d_in[18];
  const float* b_c2  = (const float*)d_in[19];

  float* out = (float*)d_out;
  float* out_hidden = out;                              // [B,H]
  float* out_logits = out + BB * HH;                    // [T,B,A]
  float* out_value  = out + BB * HH + (size_t)TB * AA;  // [T,B]

  uint8_t* base = (uint8_t*)d_ws;
  size_t off = 0;
  auto alloc = [&](size_t bytes) -> void* {
    void* p = base + off;
    off += (bytes + 255) & ~(size_t)255;
    return p;
  };
  unsigned* bar  = (unsigned*)alloc(256);
  float*   hbuf  = (float*)alloc((size_t)2 * BB * HH * 4);
  __bf16*  hbf   = (__bf16*)alloc((size_t)2 * BB * HH * 2);
  __bf16*  emb   = (__bf16*)alloc((size_t)TB * HH * 2);
  float*   gi    = (float*)alloc((size_t)TB * 3 * HH * 4);
  __bf16*  yb    = (__bf16*)alloc((size_t)TB * HH * 2);
  __bf16*  Wembt = (__bf16*)alloc((size_t)HH * OBS_D * 2);
  __bf16*  Wit   = (__bf16*)alloc((size_t)3 * HH * HH * 2);
  __bf16*  Wht   = (__bf16*)alloc((size_t)3 * HH * HH * 2);
  __bf16*  Wa1t  = (__bf16*)alloc((size_t)HH * HH * 2);
  __bf16*  Wa2t  = (__bf16*)alloc((size_t)AA * HH * 2);
  __bf16*  Wc1t  = (__bf16*)alloc((size_t)HH * HH * 2);
  __bf16*  Wc2t  = (__bf16*)alloc((size_t)16 * HH * 2);

  auto tgrid = [](int elems) { return dim3((elems + 255) / 256); };

  k_init<<<tgrid(BB * HH), 256, 0, stream>>>(bar, hbuf, hbf, hidden);
  k_transpose_bf16<<<tgrid(HH * OBS_D), 256, 0, stream>>>(W_emb, Wembt, OBS_D, HH, HH);
  k_transpose_bf16<<<tgrid(3 * HH * HH), 256, 0, stream>>>(Wi, Wit, HH, 3 * HH, 3 * HH);
  k_transpose_bf16<<<tgrid(HH * HH), 256, 0, stream>>>(Whr, Wht + 0 * HH * HH, HH, HH, HH);
  k_transpose_bf16<<<tgrid(HH * HH), 256, 0, stream>>>(Whz, Wht + 1 * HH * HH, HH, HH, HH);
  k_transpose_bf16<<<tgrid(HH * HH), 256, 0, stream>>>(Whn, Wht + 2 * HH * HH, HH, HH, HH);
  k_transpose_bf16<<<tgrid(HH * HH), 256, 0, stream>>>(W_a1, Wa1t, HH, HH, HH);
  k_transpose_bf16<<<tgrid(AA * HH), 256, 0, stream>>>(W_a2, Wa2t, HH, AA, AA);
  k_transpose_bf16<<<tgrid(HH * HH), 256, 0, stream>>>(W_c1, Wc1t, HH, HH, HH);
  k_transpose_bf16<<<tgrid(16 * HH), 256, 0, stream>>>(W_c2, Wc2t, HH, 1, 16);

  // one wave per 16-row block (A reused across all N tiles)
  k_emb<<<dim3(TB / 16 / 8), 256, 0, stream>>>(obs, Wembt, b_emb, emb);
  k_gi<<<dim3(TB / 16 / 8), 256, 0, stream>>>(emb, Wit, bi, gi);
  k_gru<<<dim3(32), 256, 0, stream>>>(hbuf, hbf, gi, dones, Wht, bhn, yb, out_hidden, bar);
  k_heads<<<dim3(TB / 16 / 8), 256, 0, stream>>>(yb, Wa1t, b_a1, Wa2t, b_a2,
                                                 Wc1t, b_c1, Wc2t, b_c2, avail,
                                                 out_logits, out_value);
}
